// MultiHeadAttention_70128226009797
// MI455X (gfx1250) — matrix operand order repaired and checked by reading, not compiled, not run on hardware
//
#include <hip/hip_runtime.h>
#include <hip/hip_bf16.h>

// ---------------------------------------------------------------------------
// MI455X (gfx1250) multi-head attention, f16 WMMA pipeline with f32 accum.
// B=2, S=4096, D=512, H=8, HD=64.
// ---------------------------------------------------------------------------

#define BDIM  2
#define SDIM  4096
#define DDIM  512
#define HDIM  8
#define HDHD  64
#define MROWS (BDIM * SDIM)   // 8192

typedef __attribute__((ext_vector_type(16))) _Float16     v16h;
typedef __attribute__((ext_vector_type(8)))  _Float16     h8;
typedef __attribute__((ext_vector_type(4)))  _Float16     h4;
typedef __attribute__((ext_vector_type(8)))  float        v8f;
typedef __attribute__((ext_vector_type(4)))  unsigned int v4u;
typedef __attribute__((ext_vector_type(4)))  int          v4i;
typedef __attribute__((ext_vector_type(8)))  int          v8i;

// Build a 16x32 A-fragment / 32x16 B-fragment register image from two
// contiguous 16-byte chunks (contiguous k = 16*(lane/16) + s, same for A and B, wave32).
static __device__ __forceinline__ v16h ldfrag(const _Float16* p0,
                                              const _Float16* p1) {
  h8 lo = *(const h8*)p0;
  h8 hi = *(const h8*)p1;
  return __builtin_shufflevector(lo, hi, 0, 1, 2, 3, 4, 5, 6, 7,
                                         8, 9, 10, 11, 12, 13, 14, 15);
}

static __device__ __forceinline__ v8f wmma32(v16h a, v16h b, v8f c) {
  // D = A(16x32 f16) * B(32x16 f16) + C(16x16 f32)
  return __builtin_amdgcn_wmma_f32_16x16x32_f16(
      /*neg_a=*/false, a, /*neg_b=*/false, b,
      /*c_mod=*/(short)0, c, /*reuse_a=*/false, /*reuse_b=*/false);
}

// ---------------------------------------------------------------------------
// Tensor Data Mover: 2D f16 tile (tile_w x tile_h elements) from global to
// LDS, with hardware padding so LDS row stride = tile_w + 8 halves.
// D# built per CDNA5 ISA 8.3/8.4.  pad_interval code 4 = 32 dwords (128B row
// for tile_w=64), pad_amount code 3 = 4 dwords (16B) -> 144B stride.
// ---------------------------------------------------------------------------
static __device__ __forceinline__ void tdm_load_tile_f16(
    unsigned int lds_byte_off, const _Float16* gsrc,
    unsigned int tensor_w, unsigned int tensor_h,
    unsigned int tile_w, unsigned int tile_h,
    unsigned int stride_elems) {
  unsigned long long ga = (unsigned long long)(uintptr_t)gsrc;
  v4u g0;
  g0[0] = 1u;                                  // count=1, no gather/restore
  g0[1] = lds_byte_off;                        // LDS dest (bytes)
  g0[2] = (unsigned int)ga;                    // global_addr[31:0]
  g0[3] = (unsigned int)((ga >> 32) & 0x01FFFFFFu) | (2u << 30);  // type=2
  v8i g1;
  g1[0] = (int)((1u << 16)      // data_size = 1 -> 2-byte elements
              | (1u << 20)      // pad_enable
              | (4u << 22)      // pad_interval: 32 dwords (128B)
              | (3u << 25));    // pad_amount: 4 dwords (16B)
  g1[1] = (int)(tensor_w << 16);                         // tensor_dim0 lo16
  g1[2] = (int)((tensor_w >> 16) | (tensor_h << 16));    // dim0 hi / dim1 lo
  g1[3] = (int)((tensor_h >> 16) | (tile_w << 16));      // dim1 hi / tile0
  g1[4] = (int)(tile_h & 0xFFFFu);                       // tile1 (tile2 = 0)
  g1[5] = (int)stride_elems;                             // dim0_stride lo32
  g1[6] = 0;                                             // stride hi / d1str
  g1[7] = 0;
  v4i z4 = {0, 0, 0, 0};
#if __clang_major__ >= 23
  v8i z8 = {0, 0, 0, 0, 0, 0, 0, 0};
  __builtin_amdgcn_tensor_load_to_lds(g0, g1, z4, z4, z8, 0);
#else
  __builtin_amdgcn_tensor_load_to_lds(g0, g1, z4, z4, 0);
#endif
}

// ---------------------------------------------------------------------------
// Kernel 1: fused projection GEMM.  out[b,h,s,hd] (f16) = A[m,:] @ W + bias
// A: fp32 [8192 x 512], W: fp32 [512 x 512], out: f16 in [B,H,S,HD] layout.
// WG = 256 threads = 8 waves; WG tile 128(M) x 64(N); wave tile 32 x 32
// (2 A-frags x 2 B-frags -> 4 WMMAs per K-step).
// ---------------------------------------------------------------------------
__global__ __launch_bounds__(256) void qkv_proj_kernel(
    const float* __restrict__ A, const float* __restrict__ W,
    const float* __restrict__ bias, _Float16* __restrict__ out) {
  __shared__ _Float16 la[128 * 40];  // A tile, row-major [m][k], stride 40
  __shared__ _Float16 lb[64 * 40];   // W tile, transposed [n][k], stride 40

  const int t    = threadIdx.x;
  const int lane = t & 31;
  const int wave = t >> 5;
  const int grp  = lane >> 4;   // half-wave
  const int lr   = lane & 15;
  const int wr   = wave & 3;    // wave row-tile (0..3) -> 32 rows each
  const int wc   = wave >> 2;   // wave col-tile (0..1) -> 32 cols each
  const int row0 = blockIdx.y * 128;
  const int col0 = blockIdx.x * 64;

  v8f acc[2][2] = {};

  for (int k0 = 0; k0 < DDIM; k0 += 32) {
    // ---- stage A tile 128x32 (fp32 -> f16) ------------------------------
    #pragma unroll
    for (int it = 0; it < 4; ++it) {
      const int idx = t + 256 * it;  // 0..1023
      const int r   = idx >> 3;      // 0..127
      const int cg  = idx & 7;       // 4 floats each
      float4 v = *(const float4*)&A[(size_t)(row0 + r) * DDIM + k0 + cg * 4];
      h4 hv = { (_Float16)v.x, (_Float16)v.y, (_Float16)v.z, (_Float16)v.w };
      *(h4*)&la[r * 40 + cg * 4] = hv;
    }
    // ---- stage W tile 32x64 transposed -> [n][k] ------------------------
    #pragma unroll
    for (int it = 0; it < 2; ++it) {
      const int idx = t + 256 * it;  // 0..511
      const int kk  = idx >> 4;      // 0..31
      const int ng  = idx & 15;      // 4 cols each
      float4 v = *(const float4*)&W[(size_t)(k0 + kk) * DDIM + col0 + ng * 4];
      lb[(ng * 4 + 0) * 40 + kk] = (_Float16)v.x;
      lb[(ng * 4 + 1) * 40 + kk] = (_Float16)v.y;
      lb[(ng * 4 + 2) * 40 + kk] = (_Float16)v.z;
      lb[(ng * 4 + 3) * 40 + kk] = (_Float16)v.w;
    }
    // ---- prefetch next K-step tiles (global_prefetch_b8) ----------------
    if (k0 + 32 < DDIM) {
      __builtin_prefetch(&A[(size_t)(row0 + (t >> 1)) * DDIM + k0 + 32 +
                            (t & 1) * 16], 0, 0);
      __builtin_prefetch(&W[(size_t)(k0 + 32 + (t >> 3)) * DDIM + col0 +
                            (t & 7) * 8], 0, 0);
    }
    __syncthreads();

    const _Float16* ar0 = &la[(32 * wr + lr) * 40];
    const _Float16* ar1 = &la[(32 * wr + 16 + lr) * 40];
    v16h af0 = ldfrag(ar0 + 16 * grp, ar0 + 16 * grp + 8);
    v16h af1 = ldfrag(ar1 + 16 * grp, ar1 + 16 * grp + 8);

    const _Float16* bp0 = &lb[(32 * wc + lr) * 40 + 16 * grp];
    const _Float16* bp1 = &lb[(32 * wc + 16 + lr) * 40 + 16 * grp];
    v16h bf0 = ldfrag(bp0, bp0 + 8);
    v16h bf1 = ldfrag(bp1, bp1 + 8);

    acc[0][0] = wmma32(af0, bf0, acc[0][0]);
    acc[0][1] = wmma32(af0, bf1, acc[0][1]);
    acc[1][0] = wmma32(af1, bf0, acc[1][0]);
    acc[1][1] = wmma32(af1, bf1, acc[1][1]);

    __syncthreads();
  }

  // ---- epilogue: bias + convert, scatter to [B,H,S,HD] ------------------
  #pragma unroll
  for (int ri = 0; ri < 2; ++ri) {
    #pragma unroll
    for (int ci = 0; ci < 2; ++ci) {
      const int gn = col0 + 32 * wc + 16 * ci + lr;
      const int hh = gn >> 6;
      const int hd = gn & 63;
      const float bv = bias[gn];
      #pragma unroll
      for (int j = 0; j < 8; ++j) {
        const int gm = row0 + 32 * wr + 16 * ri + j + 8 * grp;
        const int bb = gm >> 12;
        const int ss = gm & 4095;
        out[(((size_t)(bb * HDIM + hh)) * SDIM + ss) * HDHD + hd] =
            (_Float16)(acc[ri][ci][j] + bv);
      }
    }
  }
}

// ---------------------------------------------------------------------------
// Kernel 2: flash attention.  q,k,v: f16 [B,H,S,64]; o: f16 [B,S,512].
// WG = 128 threads = 4 waves; each wave owns 16 query rows (WG: 64 rows).
// K tile DMA'd by the Tensor Data Mover (wave 0) with HW padding; V tile
// staged transposed by ds_store; online softmax; all matmuls via WMMA.
// ---------------------------------------------------------------------------
__global__ __launch_bounds__(128) void attn_kernel(
    const _Float16* __restrict__ q, const _Float16* __restrict__ k,
    const _Float16* __restrict__ v, _Float16* __restrict__ o) {
  __shared__ _Float16 lk[32 * 72];       // K tile [key][hd], stride 72 (TDM pad)
  __shared__ _Float16 lvt[64 * 40];      // V tile transposed [hd][key]
  __shared__ _Float16 lp[4 * 16 * 40];   // per-wave P scratch [row][key]

  const int t    = threadIdx.x;
  const int lane = t & 31;
  const int wave = t >> 5;
  const int grp  = lane >> 4;
  const int lr   = lane & 15;
  const int bb   = blockIdx.z;
  const int hh   = blockIdx.y;
  const int q0   = blockIdx.x * 64 + wave * 16;

  const size_t head = ((size_t)(bb * HDIM + hh)) * SDIM * HDHD;
  const _Float16* qb = q + head;
  const _Float16* kb = k + head;
  const _Float16* vb = v + head;

  // Q fragments for this wave's 16 rows (HD=64 -> 2 K-chunks of 32)
  const _Float16* qr = qb + (size_t)(q0 + lr) * HDHD;
  const v16h aq0 = ldfrag(qr + 16 * grp, qr + 16 * grp + 8);
  const v16h aq1 = ldfrag(qr + 32 + 16 * grp, qr + 40 + 16 * grp);

  float m[8], l[8];
  #pragma unroll
  for (int j = 0; j < 8; ++j) { m[j] = -3.0e38f; l[j] = 0.0f; }
  v8f oa[4] = {};   // 16 rows x 64 hd output accumulator (4 N-subtiles)

  const unsigned int lk_off = (unsigned int)(uintptr_t)&lk[0];

  for (int kb0 = 0; kb0 < SDIM; kb0 += 32) {
    // ---- K tile 32x64 via Tensor Data Mover (wave 0 issues the DMA) -----
    if (wave == 0) {
      tdm_load_tile_f16(lk_off, kb + (size_t)kb0 * HDHD,
                        /*tensor_w=*/HDHD, /*tensor_h=*/SDIM,
                        /*tile_w=*/HDHD, /*tile_h=*/32,
                        /*stride=*/HDHD);
    }
    // ---- stage V tile transposed [hd][key] ------------------------------
    #pragma unroll
    for (int it = 0; it < 2; ++it) {
      const int idx = t + 128 * it;  // 0..255
      const int key = idx >> 3;      // 0..31
      const int cg  = idx & 7;       // 8 halves each
      h8 vvv = *(const h8*)&vb[(size_t)(kb0 + key) * HDHD + cg * 8];
      #pragma unroll
      for (int e = 0; e < 8; ++e) lvt[(cg * 8 + e) * 40 + key] = vvv[e];
    }
    // ---- prefetch next K/V tiles (each tile is 4KB contiguous) ----------
    if (kb0 + 32 < SDIM) {
      const size_t nxt = (size_t)(kb0 + 32) * HDHD;
      __builtin_prefetch(&kb[nxt + t * 16], 0, 0);
      __builtin_prefetch(&vb[nxt + t * 16], 0, 0);
    }
    __builtin_amdgcn_s_wait_tensorcnt(0);   // TDM done (wave 0; others trivial)
    __syncthreads();

    // ---- scores S = Q @ K^T (two 16-key subtiles, K-dim = HD = 64) ------
    v8f z = {};
    const _Float16* kr0 = &lk[lr * 72];
    const _Float16* kr1 = &lk[(16 + lr) * 72];
    v8f s0 = wmma32(aq0, ldfrag(kr0 + 16 * grp, kr0 + 16 * grp + 8), z);
    s0     = wmma32(aq1, ldfrag(kr0 + 32 + 16 * grp, kr0 + 40 + 16 * grp), s0);
    v8f s1 = wmma32(aq0, ldfrag(kr1 + 16 * grp, kr1 + 16 * grp + 8), z);
    s1     = wmma32(aq1, ldfrag(kr1 + 32 + 16 * grp, kr1 + 40 + 16 * grp), s1);

    // ---- online softmax (row r = j + 8*grp lives on 16 lanes) -----------
    _Float16* myp = &lp[wave * 16 * 40];
    #pragma unroll
    for (int j = 0; j < 8; ++j) {
      const float a0 = s0[j] * 0.125f;   // 1/sqrt(64)
      const float a1 = s1[j] * 0.125f;
      float mx = fmaxf(a0, a1);
      #pragma unroll
      for (int off = 1; off < 16; off <<= 1)
        mx = fmaxf(mx, __shfl_xor(mx, off, 32));
      const float mn   = fmaxf(m[j], mx);
      const float corr = __expf(m[j] - mn);
      const float p0   = __expf(a0 - mn);
      const float p1   = __expf(a1 - mn);
      float rs = p0 + p1;
      #pragma unroll
      for (int off = 1; off < 16; off <<= 1)
        rs += __shfl_xor(rs, off, 32);
      l[j] = l[j] * corr + rs;
      m[j] = mn;
      #pragma unroll
      for (int nt = 0; nt < 4; ++nt) oa[nt][j] *= corr;
      const int prow = (j + 8 * grp) * 40;
      myp[prow + lr]      = (_Float16)p0;
      myp[prow + 16 + lr] = (_Float16)p1;
    }
    // LDS write -> cross-lane read within the same wave: drain DS pipe.
    asm volatile("s_wait_dscnt 0x0" ::: "memory");

    // ---- O += P @ V -----------------------------------------------------
    const _Float16* pr = &myp[lr * 40];
    const v16h ap = ldfrag(pr + 16 * grp, pr + 16 * grp + 8);
    #pragma unroll
    for (int nt = 0; nt < 4; ++nt) {
      const _Float16* vr = &lvt[(16 * nt + lr) * 40 + 16 * grp];
      oa[nt] = wmma32(ap, ldfrag(vr, vr + 8), oa[nt]);
    }
    __syncthreads();
  }

  // ---- normalize and store in [B,S,D] layout (heads re-merged) ----------
  _Float16* ob = o + ((size_t)bb * SDIM) * DDIM + (size_t)hh * HDHD;
  #pragma unroll
  for (int j = 0; j < 8; ++j) {
    const float inv = 1.0f / l[j];
    const int ss = q0 + j + 8 * grp;
    _Float16* orow = ob + (size_t)ss * DDIM;
    orow[lr]      = (_Float16)(oa[0][j] * inv);
    orow[16 + lr] = (_Float16)(oa[1][j] * inv);
    orow[32 + lr] = (_Float16)(oa[2][j] * inv);
    orow[48 + lr] = (_Float16)(oa[3][j] * inv);
  }
}

// ---------------------------------------------------------------------------
// Kernel 3: output projection.  out(fp32) = O(f16)[8192x512] @ Wp + bp
// A fragments straight from global (already f16); Wp staged transposed in
// LDS.  Wave tile 32x32 (4 WMMAs per K-step).
// ---------------------------------------------------------------------------
__global__ __launch_bounds__(256) void out_proj_kernel(
    const _Float16* __restrict__ O, const float* __restrict__ W,
    const float* __restrict__ bias, float* __restrict__ out) {
  __shared__ _Float16 lb[64 * 40];   // W tile transposed [n][k], stride 40

  const int t    = threadIdx.x;
  const int lane = t & 31;
  const int wave = t >> 5;
  const int grp  = lane >> 4;
  const int lr   = lane & 15;
  const int wr   = wave & 3;
  const int wc   = wave >> 2;
  const int row0 = blockIdx.y * 128;
  const int col0 = blockIdx.x * 64;

  v8f acc[2][2] = {};

  const _Float16* arow0 = O + (size_t)(row0 + 32 * wr + lr) * DDIM;
  const _Float16* arow1 = O + (size_t)(row0 + 32 * wr + 16 + lr) * DDIM;

  for (int k0 = 0; k0 < DDIM; k0 += 32) {
    #pragma unroll
    for (int it = 0; it < 2; ++it) {
      const int idx = t + 256 * it;
      const int kk  = idx >> 4;
      const int ng  = idx & 15;
      float4 v = *(const float4*)&W[(size_t)(k0 + kk) * DDIM + col0 + ng * 4];
      lb[(ng * 4 + 0) * 40 + kk] = (_Float16)v.x;
      lb[(ng * 4 + 1) * 40 + kk] = (_Float16)v.y;
      lb[(ng * 4 + 2) * 40 + kk] = (_Float16)v.z;
      lb[(ng * 4 + 3) * 40 + kk] = (_Float16)v.w;
    }
    if (k0 + 32 < DDIM) {
      __builtin_prefetch(&W[(size_t)(k0 + 32 + (t >> 3)) * DDIM + col0 +
                            (t & 7) * 8], 0, 0);
    }
    __syncthreads();

    v16h af0 = ldfrag(arow0 + k0 + 16 * grp, arow0 + k0 + 16 * grp + 8);
    v16h af1 = ldfrag(arow1 + k0 + 16 * grp, arow1 + k0 + 16 * grp + 8);

    const _Float16* bp0 = &lb[(32 * wc + lr) * 40 + 16 * grp];
    const _Float16* bp1 = &lb[(32 * wc + 16 + lr) * 40 + 16 * grp];
    v16h bf0 = ldfrag(bp0, bp0 + 8);
    v16h bf1 = ldfrag(bp1, bp1 + 8);

    acc[0][0] = wmma32(af0, bf0, acc[0][0]);
    acc[0][1] = wmma32(af0, bf1, acc[0][1]);
    acc[1][0] = wmma32(af1, bf0, acc[1][0]);
    acc[1][1] = wmma32(af1, bf1, acc[1][1]);

    __syncthreads();
  }

  #pragma unroll
  for (int ri = 0; ri < 2; ++ri) {
    #pragma unroll
    for (int ci = 0; ci < 2; ++ci) {
      const int gn = col0 + 32 * wc + 16 * ci + lr;
      const float bv = bias[gn];
      #pragma unroll
      for (int j = 0; j < 8; ++j) {
        const int gm = row0 + 32 * wr + 16 * ri + j + 8 * grp;
        out[(size_t)gm * DDIM + gn] = acc[ri][ci][j] + bv;
      }
    }
  }
}

// ---------------------------------------------------------------------------
// Host launch.  Workspace layout (f16 elements):
//   [0 .. 4M)      Q   [B,H,S,64]
//   [4M .. 8M)     K
//   [8M .. 12M)    V
//   [12M .. 16M)   O   [B,S,512]     (16M halves = 32 MB)
// ---------------------------------------------------------------------------
extern "C" void kernel_launch(void* const* d_in, const int* in_sizes, int n_in,
                              void* d_out, int out_size, void* d_ws,
                              size_t ws_size, hipStream_t stream) {
  (void)in_sizes; (void)n_in; (void)out_size; (void)ws_size;

  const float* x  = (const float*)d_in[0];
  const float* y  = (const float*)d_in[1];
  const float* z  = (const float*)d_in[2];
  const float* Wq = (const float*)d_in[3];
  const float* bq = (const float*)d_in[4];
  const float* Wk = (const float*)d_in[5];
  const float* bk = (const float*)d_in[6];
  const float* Wv = (const float*)d_in[7];
  const float* bv = (const float*)d_in[8];
  const float* Wp = (const float*)d_in[9];
  const float* bp = (const float*)d_in[10];
  float* out = (float*)d_out;

  const size_t QKV = (size_t)BDIM * HDIM * SDIM * HDHD;  // 4,194,304 halves
  _Float16* ws = (_Float16*)d_ws;
  _Float16* qw = ws;
  _Float16* kw = ws + QKV;
  _Float16* vw = ws + 2 * QKV;
  _Float16* ow = ws + 3 * QKV;

  const dim3 gproj(DDIM / 64, MROWS / 128);   // 8 x 64
  qkv_proj_kernel<<<gproj, 256, 0, stream>>>(x, Wq, bq, qw);
  qkv_proj_kernel<<<gproj, 256, 0, stream>>>(y, Wk, bk, kw);
  qkv_proj_kernel<<<gproj, 256, 0, stream>>>(z, Wv, bv, vw);

  const dim3 gattn(SDIM / 64, HDIM, BDIM);    // 64 x 8 x 2
  attn_kernel<<<gattn, 128, 0, stream>>>(qw, kw, vw, ow);

  out_proj_kernel<<<gproj, 256, 0, stream>>>(ow, Wp, bp, out);
}
